// MultiHeadAttention_80272938762821
// MI455X (gfx1250) — matrix operand order repaired and checked by reading, not compiled, not run on hardware
//
#include <hip/hip_runtime.h>
#include <hip/hip_bf16.h>

#define NB  2
#define SEQ 2048
#define EMB 1024
#define NH  16
#define HD  64

typedef __bf16 bf16_t;
typedef __attribute__((ext_vector_type(16))) __bf16 v16bf;
typedef __attribute__((ext_vector_type(8)))  __bf16 v8bf;
typedef __attribute__((ext_vector_type(8)))  float  v8f;

union FragU { v16bf v; v8bf h[2]; };

__device__ __forceinline__ v8f wmma_bf16(v16bf a, v16bf b, v8f c) {
  return __builtin_amdgcn_wmma_f32_16x16x32_bf16(false, a, false, b, (short)0, c, false, false);
}

// ---- CDNA5 async global->LDS copy (ASYNCcnt tracked) ------------------------
__device__ __forceinline__ void async_ld_b128(uint32_t lds_off, const bf16_t* g) {
  asm volatile("global_load_async_to_lds_b128 %0, %1, off"
               :: "v"(lds_off), "v"(g) : "memory");
}
#if __has_builtin(__builtin_amdgcn_s_wait_asynccnt)
#define WAIT_ASYNC(n) __builtin_amdgcn_s_wait_asynccnt(n)
#else
#define WAIT_ASYNC(n) asm volatile("s_wait_asynccnt %0" :: "i"(n) : "memory")
#endif

__device__ __forceinline__ uint32_t lds_off_u32(const void* p) {
  return (uint32_t)(uintptr_t)p;  // flat LDS aperture keeps the offset in [31:0]
}

// A-matrix fragment (16x32 tile, row-major bf16 source, stride in elements).
__device__ __forceinline__ v16bf load_frag_a(const bf16_t* base, int stride) {
  const int lane = threadIdx.x & 31;
  const bf16_t* p = base + (lane & 15) * stride + ((lane >> 4) << 3);
  FragU u;
  u.h[0] = *(const v8bf*)(p);
  u.h[1] = *(const v8bf*)(p + 16);
  return u.v;
}

// Same A layout but converting from fp32 source on the fly.
__device__ __forceinline__ v16bf load_frag_a_f32(const float* base, int stride) {
  const int lane = threadIdx.x & 31;
  const float* p = base + (lane & 15) * stride + ((lane >> 4) << 3);
  v16bf f;
#pragma unroll
  for (int i = 0; i < 8; ++i) f[i] = (bf16_t)p[i];
#pragma unroll
  for (int i = 0; i < 8; ++i) f[i + 8] = (bf16_t)p[i + 16];
  return f;
}

// B-matrix fragment (32x16 tile) from a row-major *transposed* source.
__device__ __forceinline__ v16bf load_frag_b(const bf16_t* baseT, int stride) {
  const int lane = threadIdx.x & 31;
  const bf16_t* p = baseT + (lane & 15) * stride + ((lane >> 4) << 3);
  FragU u;
  u.h[0] = *(const v8bf*)(p);
  u.h[1] = *(const v8bf*)(p + 16);
  return u.v;
}

// ---------------------------------------------------------------- cvt weights
__global__ void cvt_f32_bf16_kernel(const float* __restrict__ src,
                                    bf16_t* __restrict__ dst, int n) {
  for (int i = blockIdx.x * blockDim.x + threadIdx.x; i < n;
       i += gridDim.x * blockDim.x)
    dst[i] = (bf16_t)src[i];
}

// ------------------------------------------------- Q/K/V projection (bf16 out)
// Y = X @ W^T + bias ; wave owns 16 rows x 64 cols (one head's slice).
// Explicit ping-pong software pipeline: zero register-copy rotation.
__global__ void __launch_bounds__(128)
qkv_proj_kernel(const float* __restrict__ X, const bf16_t* __restrict__ W,
                const float* __restrict__ bias, bf16_t* __restrict__ dst,
                int vt_mode) {
  const int task  = blockIdx.x * (blockDim.x >> 5) + (threadIdx.x >> 5);
  const int mtile = task >> 4;
  const int h     = task & 15;
  const int lane  = threadIdx.x & 31;
  const int nl = lane & 15, hl = lane >> 4;

  v8f acc[4] = {};
  const float*  a_base = X + (size_t)(mtile * 16) * EMB;
  const bf16_t* b_base = W + (size_t)(h * 64) * EMB;

  v16bf aA = load_frag_a_f32(a_base, EMB);
  v16bf wA[4];
#pragma unroll
  for (int j = 0; j < 4; ++j) wA[j] = load_frag_b(b_base + (size_t)(j * 16) * EMB, EMB);

  for (int k = 0; k < EMB - 64; k += 64) {
    v16bf aB = load_frag_a_f32(a_base + k + 32, EMB);
    v16bf wB[4];
#pragma unroll
    for (int j = 0; j < 4; ++j)
      wB[j] = load_frag_b(b_base + (size_t)(j * 16) * EMB + k + 32, EMB);
#pragma unroll
    for (int j = 0; j < 4; ++j) acc[j] = wmma_bf16(aA, wA[j], acc[j]);

    aA = load_frag_a_f32(a_base + k + 64, EMB);
#pragma unroll
    for (int j = 0; j < 4; ++j)
      wA[j] = load_frag_b(b_base + (size_t)(j * 16) * EMB + k + 64, EMB);
#pragma unroll
    for (int j = 0; j < 4; ++j) acc[j] = wmma_bf16(aB, wB[j], acc[j]);
  }
  {  // epilogue: k = EMB-64 and k = EMB-32
    v16bf aB = load_frag_a_f32(a_base + EMB - 32, EMB);
    v16bf wB[4];
#pragma unroll
    for (int j = 0; j < 4; ++j)
      wB[j] = load_frag_b(b_base + (size_t)(j * 16) * EMB + EMB - 32, EMB);
#pragma unroll
    for (int j = 0; j < 4; ++j) acc[j] = wmma_bf16(aA, wA[j], acc[j]);
#pragma unroll
    for (int j = 0; j < 4; ++j) acc[j] = wmma_bf16(aB, wB[j], acc[j]);
  }

#pragma unroll
  for (int j = 0; j < 4; ++j) {
#pragma unroll
    for (int r = 0; r < 8; ++r) {
      int mg = mtile * 16 + r + 8 * hl;
      int bb = mg >> 11;
      int s  = mg & (SEQ - 1);
      int d  = j * 16 + nl;
      float v = acc[j][r] + bias[h * 64 + d];
      size_t idx;
      if (vt_mode) idx = ((size_t)(bb * NH + h) * HD + d) * SEQ + s;
      else         idx = ((size_t)(bb * NH + h) * SEQ + s) * HD + d;
      dst[idx] = (bf16_t)v;
    }
  }
}

// ------------------------------------------------------------------ attention
// Block = 4 waves sharing one (b,h); each wave owns one 16-query tile.
// K/V chunks staged to LDS once per block via async copies, double-buffered.
__global__ void __launch_bounds__(128)
attn_kernel(const bf16_t* __restrict__ Q, const bf16_t* __restrict__ K,
            const bf16_t* __restrict__ Vt, const int* __restrict__ mask,
            float* __restrict__ attn_out, bf16_t* __restrict__ O) {
  __shared__ bf16_t kbuf[2][32 * 64];   // 32 keys x 64 d
  __shared__ bf16_t vbuf[2][64 * 32];   // 64 d x 32 t
  __shared__ bf16_t pbuf[4][16 * 32];   // per-wave P tile

  const int wave = threadIdx.x >> 5;
  const int bh   = blockIdx.x >> 5;                // 0..31 = b*16+h
  const int qt   = (blockIdx.x & 31) * 4 + wave;   // 0..127
  const int lane = threadIdx.x & 31;
  const int nl = lane & 15, hl = lane >> 4;
  const int qbase = qt * 16;
  const int tid = threadIdx.x;

  const bf16_t* Qh = Q  + (size_t)bh * SEQ * HD;
  const bf16_t* Kh = K  + (size_t)bh * SEQ * HD;
  const bf16_t* Vh = Vt + (size_t)bh * HD * SEQ;

  const v16bf qf0 = load_frag_a(Qh + (size_t)qbase * HD, HD);
  const v16bf qf1 = load_frag_a(Qh + (size_t)qbase * HD + 32, HD);

  // block-cooperative async staging helpers (2 b128 ops/thread each)
  auto stage_k32 = [&](int tc, bf16_t* dst) {
#pragma unroll
    for (int i = 0; i < 2; ++i) {
      int seg = tid + i * 128;            // 0..255
      int row = seg >> 3, s = seg & 7;    // 32 rows x 8 segs of 16B
      async_ld_b128(lds_off_u32(dst + row * 64 + s * 8),
                    Kh + (size_t)(tc + row) * HD + s * 8);
    }
  };
  auto stage_v32 = [&](int tc, bf16_t* dst) {
#pragma unroll
    for (int i = 0; i < 2; ++i) {
      int seg = tid + i * 128;
      int row = seg >> 2, s = seg & 3;    // 64 rows x 4 segs of 16B
      async_ld_b128(lds_off_u32(dst + row * 32 + s * 8),
                    Vh + (size_t)row * SEQ + tc + s * 8);
    }
  };

  float mrow[8], lrow[8];
#pragma unroll
  for (int r = 0; r < 8; ++r) { mrow[r] = -1e30f; lrow[r] = 0.0f; }

  // -------- sweep 1: online softmax statistics --------
  stage_k32(0, kbuf[0]);
  for (int ch = 0; ch < 64; ++ch) {
    const int tc = ch * 32;
    if (ch + 1 < 64) { stage_k32(tc + 32, kbuf[(ch + 1) & 1]); WAIT_ASYNC(2); }
    else             { WAIT_ASYNC(0); }
    __syncthreads();
    const bf16_t* kb = kbuf[ch & 1];
#pragma unroll
    for (int sub = 0; sub < 2; ++sub) {
      const int t = tc + sub * 16;
      v16bf kf0 = load_frag_b(kb + sub * 16 * 64, 64);
      v16bf kf1 = load_frag_b(kb + sub * 16 * 64 + 32, 64);
      v8f c = {};
      c = wmma_bf16(qf0, kf0, c);
      c = wmma_bf16(qf1, kf1, c);
#pragma unroll
      for (int r = 0; r < 8; ++r) {
        int mq = qbase + r + 8 * hl;
        float s = c[r] * 0.125f;                       // 1/sqrt(64)
        if (mask[mq * SEQ + t + nl] == 0) s = -1e30f;
        float tm = s;
        tm = fmaxf(tm, __shfl_xor(tm, 1, 32));
        tm = fmaxf(tm, __shfl_xor(tm, 2, 32));
        tm = fmaxf(tm, __shfl_xor(tm, 4, 32));
        tm = fmaxf(tm, __shfl_xor(tm, 8, 32));
        float nm = fmaxf(mrow[r], tm);
        float p = __expf(s - nm);
        float ts = p;
        ts += __shfl_xor(ts, 1, 32);
        ts += __shfl_xor(ts, 2, 32);
        ts += __shfl_xor(ts, 4, 32);
        ts += __shfl_xor(ts, 8, 32);
        lrow[r] = lrow[r] * __expf(mrow[r] - nm) + ts;
        mrow[r] = nm;
      }
    }
    __syncthreads();
  }
  float invl[8];
#pragma unroll
  for (int r = 0; r < 8; ++r) invl[r] = 1.0f / lrow[r];

  // -------- sweep 2: write attn probs (non-temporal), accumulate O = P @ V ---
  v8f o[4] = {};
  stage_k32(0, kbuf[0]);
  stage_v32(0, vbuf[0]);
  for (int ch = 0; ch < 64; ++ch) {
    const int tc = ch * 32;
    if (ch + 1 < 64) {
      stage_k32(tc + 32, kbuf[(ch + 1) & 1]);
      stage_v32(tc + 32, vbuf[(ch + 1) & 1]);
      WAIT_ASYNC(4);
    } else {
      WAIT_ASYNC(0);
    }
    __syncthreads();
    const bf16_t* kb = kbuf[ch & 1];
    const bf16_t* vb = vbuf[ch & 1];
#pragma unroll
    for (int sub = 0; sub < 2; ++sub) {
      const int t = tc + sub * 16;
      v16bf kf0 = load_frag_b(kb + sub * 16 * 64, 64);
      v16bf kf1 = load_frag_b(kb + sub * 16 * 64 + 32, 64);
      v8f c = {};
      c = wmma_bf16(qf0, kf0, c);
      c = wmma_bf16(qf1, kf1, c);
#pragma unroll
      for (int r = 0; r < 8; ++r) {
        int mq = qbase + r + 8 * hl;
        float s = c[r] * 0.125f;
        if (mask[mq * SEQ + t + nl] == 0) s = -1e30f;
        float p = __expf(s - mrow[r]) * invl[r];
        __builtin_nontemporal_store(p, &attn_out[((size_t)bh * SEQ + mq) * SEQ + t + nl]);
        pbuf[wave][(r + 8 * hl) * 32 + sub * 16 + nl] = (bf16_t)p;
      }
    }
    // same-wave LDS store->load is in order; C-layout -> A-layout via LDS
    v16bf pf = load_frag_a(&pbuf[wave][0], 32);
#pragma unroll
    for (int j = 0; j < 4; ++j) {
      v16bf vf = load_frag_b(vb + (j * 16) * 32, 32);
      o[j] = wmma_bf16(pf, vf, o[j]);
    }
    __syncthreads();
  }

  const int bb = bh >> 4, h = bh & 15;
#pragma unroll
  for (int j = 0; j < 4; ++j) {
#pragma unroll
    for (int r = 0; r < 8; ++r) {
      int mq = qbase + r + 8 * hl;
      int e  = h * 64 + j * 16 + nl;
      O[((size_t)bb * SEQ + mq) * EMB + e] = (bf16_t)o[j][r];
    }
  }
}

// ------------------------------------------------------- output projection
__global__ void __launch_bounds__(128)
oproj_kernel(const bf16_t* __restrict__ X, const bf16_t* __restrict__ W,
             const float* __restrict__ bias, float* __restrict__ out) {
  const int task  = blockIdx.x * (blockDim.x >> 5) + (threadIdx.x >> 5);
  const int mtile = task >> 4;
  const int strip = task & 15;
  const int lane  = threadIdx.x & 31;
  const int nl = lane & 15, hl = lane >> 4;

  v8f acc[4] = {};
  const bf16_t* a_base = X + (size_t)(mtile * 16) * EMB;
  const bf16_t* b_base = W + (size_t)(strip * 64) * EMB;

  v16bf aA = load_frag_a(a_base, EMB);
  v16bf wA[4];
#pragma unroll
  for (int j = 0; j < 4; ++j) wA[j] = load_frag_b(b_base + (size_t)(j * 16) * EMB, EMB);

  for (int k = 0; k < EMB - 64; k += 64) {
    v16bf aB = load_frag_a(a_base + k + 32, EMB);
    v16bf wB[4];
#pragma unroll
    for (int j = 0; j < 4; ++j)
      wB[j] = load_frag_b(b_base + (size_t)(j * 16) * EMB + k + 32, EMB);
#pragma unroll
    for (int j = 0; j < 4; ++j) acc[j] = wmma_bf16(aA, wA[j], acc[j]);

    aA = load_frag_a(a_base + k + 64, EMB);
#pragma unroll
    for (int j = 0; j < 4; ++j)
      wA[j] = load_frag_b(b_base + (size_t)(j * 16) * EMB + k + 64, EMB);
#pragma unroll
    for (int j = 0; j < 4; ++j) acc[j] = wmma_bf16(aB, wB[j], acc[j]);
  }
  {
    v16bf aB = load_frag_a(a_base + EMB - 32, EMB);
    v16bf wB[4];
#pragma unroll
    for (int j = 0; j < 4; ++j)
      wB[j] = load_frag_b(b_base + (size_t)(j * 16) * EMB + EMB - 32, EMB);
#pragma unroll
    for (int j = 0; j < 4; ++j) acc[j] = wmma_bf16(aA, wA[j], acc[j]);
#pragma unroll
    for (int j = 0; j < 4; ++j) acc[j] = wmma_bf16(aB, wB[j], acc[j]);
  }

#pragma unroll
  for (int j = 0; j < 4; ++j) {
#pragma unroll
    for (int r = 0; r < 8; ++r) {
      int mg   = mtile * 16 + r + 8 * hl;
      int ncol = strip * 64 + j * 16 + nl;
      out[(size_t)mg * EMB + ncol] = acc[j][r] + bias[ncol];
    }
  }
}

// ---------------------------------------------------------------- launcher
extern "C" void kernel_launch(void* const* d_in, const int* in_sizes, int n_in,
                              void* d_out, int out_size, void* d_ws, size_t ws_size,
                              hipStream_t stream) {
  const float* q_data = (const float*)d_in[0];
  const float* k_data = (const float*)d_in[1];
  const float* v_data = (const float*)d_in[2];
  const float* Wq_w = (const float*)d_in[3];
  const float* Wq_b = (const float*)d_in[4];
  const float* Wk_w = (const float*)d_in[5];
  const float* Wk_b = (const float*)d_in[6];
  const float* Wv_w = (const float*)d_in[7];
  const float* Wv_b = (const float*)d_in[8];
  const float* Wo_w = (const float*)d_in[9];
  const float* Wo_b = (const float*)d_in[10];
  const int*   mask = (const int*)d_in[11];

  char* ws = (char*)d_ws;
  const size_t MB = 1024 * 1024;
  bf16_t* Wqb = (bf16_t*)(ws + 0 * MB);
  bf16_t* Wkb = (bf16_t*)(ws + 2 * MB);
  bf16_t* Wvb = (bf16_t*)(ws + 4 * MB);
  bf16_t* Wob = (bf16_t*)(ws + 6 * MB);
  bf16_t* Qb  = (bf16_t*)(ws + 8 * MB);   // [N,H,S,D]
  bf16_t* Kb  = (bf16_t*)(ws + 16 * MB);  // [N,H,S,D]
  bf16_t* Vtb = (bf16_t*)(ws + 24 * MB);  // [N,H,D,S]
  bf16_t* Ob  = (bf16_t*)(ws + 32 * MB);  // [N,S,E]

  float* out  = (float*)d_out;
  float* attn = out + (size_t)NB * SEQ * EMB;

  const int wN = EMB * EMB;
  cvt_f32_bf16_kernel<<<512, 256, 0, stream>>>(Wq_w, Wqb, wN);
  cvt_f32_bf16_kernel<<<512, 256, 0, stream>>>(Wk_w, Wkb, wN);
  cvt_f32_bf16_kernel<<<512, 256, 0, stream>>>(Wv_w, Wvb, wN);
  cvt_f32_bf16_kernel<<<512, 256, 0, stream>>>(Wo_w, Wob, wN);

  qkv_proj_kernel<<<1024, 128, 0, stream>>>(q_data, Wqb, Wq_b, Qb, 0);
  qkv_proj_kernel<<<1024, 128, 0, stream>>>(k_data, Wkb, Wk_b, Kb, 0);
  qkv_proj_kernel<<<1024, 128, 0, stream>>>(v_data, Wvb, Wv_b, Vtb, 1);

  attn_kernel<<<1024, 128, 0, stream>>>(Qb, Kb, Vtb, mask, attn, Ob);

  oproj_kernel<<<1024, 128, 0, stream>>>(Ob, Wob, Wo_b, out);
}
